// GCNInteraction_64888365908354
// MI455X (gfx1250) — matrix operand order repaired and checked by reading, not compiled, not run on hardware
//
#include <hip/hip_runtime.h>

// Problem constants (from reference): B=8, N=1000, M=64, G=50, F=128
#define NB 8
#define NATOM 1000
#define NM 64
#define NG 50
#define NF 128
#define ROWS_TOT (NB * NATOM)          // 8000
#define OUT_ELEMS (ROWS_TOT * NF)      // 1,024,000 (out region of d_out)
// attn region of d_out starts at OUT_ELEMS, holds ROWS_TOT*NM = 512,000 floats

// ---------- vector types for WMMA ----------
typedef __attribute__((ext_vector_type(16))) __bf16 v16bf;
typedef __attribute__((ext_vector_type(8)))  __bf16 v8bf;
typedef __attribute__((ext_vector_type(8)))  float  v8f;

__device__ __forceinline__ unsigned short f2bf(float f) {
  unsigned int u = __float_as_uint(f);
  u += 0x7fffu + ((u >> 16) & 1u);            // round-to-nearest-even
  return (unsigned short)(u >> 16);
}

__device__ __forceinline__ v8f vzero8() {
  v8f z;
#pragma unroll
  for (int i = 0; i < 8; ++i) z[i] = 0.f;
  return z;
}

// ---------- CDNA5 async global->LDS copy (ASYNCcnt-tracked, per-lane addressed) ----------
// global_load_async_to_lds_b128: VDST = LDS byte address VGPR, VADDR = 64-bit global addr.
__device__ __forceinline__ void async_g2l_b128(void* ldsPtr, const void* gPtr) {
  unsigned lo = (unsigned)(unsigned long long)ldsPtr;           // flat addr low 32 = LDS offset
  unsigned long long ga = (unsigned long long)gPtr;
  asm volatile("global_load_async_to_lds_b128 %0, %1, off"
               :: "v"(lo), "v"(ga)
               : "memory");
}
__device__ __forceinline__ void wait_asynccnt0() {
  asm volatile("s_wait_asynccnt 0x0" ::: "memory");
}

// A-fragment (16x32 bf16, M x K) from LDS row-major [rows][strideK] (bf16 bits).
// ISA layout: lane L (<16) row=L holds K {0..7, 16..23}; lane L+16 same row,
// K {8..15, 24..31}. Two contiguous 16-byte runs per lane.
__device__ __forceinline__ v16bf load_a_frag(const unsigned short* lds, int rowBase,
                                             int strideK, int kb, int lane) {
  int row  = rowBase + (lane & 15);
  int half = lane >> 4;
  const unsigned short* p = lds + row * strideK + kb * 32 + half * 8;
  v8bf lo = *(const v8bf*)(p);
  v8bf hi = *(const v8bf*)(p + 16);
  v16bf a;
#pragma unroll
  for (int i = 0; i < 8; ++i) { a[i] = lo[i]; a[i + 8] = hi[i]; }
  return a;
}

// B-fragment (32x16 bf16, K x N) from pre-swizzled weights: tile-major, then
// 32 lanes x 16 contiguous bf16 (32B aligned) -> single v16bf load per lane.
__device__ __forceinline__ v16bf load_b_frag(const unsigned short* swz, int numNb,
                                             int kb, int nb, int lane) {
  const v16bf* p = (const v16bf*)(swz + ((size_t)(((kb * numNb) + nb) * 32 + lane) << 4));
  return *p;
}

#define WMMA_BF16(A, Bf, C) \
  __builtin_amdgcn_wmma_f32_16x16x32_bf16(false, (A), false, (Bf), (short)0, (C), false, false)

// ---------- weight pre-swizzle: f32 [K x Ncols] row-major -> bf16 fragment order ----------
__global__ void swizzle_w_kernel(const float* __restrict__ W, unsigned short* __restrict__ out,
                                 int K, int Kreal, int Ncols) {
  int idx = blockIdx.x * blockDim.x + threadIdx.x;
  int total = K * Ncols;
  if (idx >= total) return;
  int numNb = Ncols >> 4;
  int t    = idx >> 9;          // 512 elems per 32x16 tile
  int r    = idx & 511;
  int lane = r >> 4;
  int e    = r & 15;
  int kb   = t / numNb;
  int nb   = t % numNb;
  int col  = nb * 16 + (lane & 15);
  int k    = kb * 32 + (lane >> 4) * 8 + (e & 7) + (e >> 3) * 16;
  float v  = (k < Kreal) ? W[k * Ncols + col] : 0.f;
  out[idx] = f2bf(v);
}

// ---------- dense tile kernel: Y = X@Wa (+ba) [tanh; @Wb + bb] over 64-row tiles ----------
template <bool TWO_LAYER>
__global__ void dense_tile_kernel(const float* __restrict__ X,
                                  const unsigned short* __restrict__ sWa,
                                  const float* __restrict__ ba,
                                  const unsigned short* __restrict__ sWb,
                                  const float* __restrict__ bb,
                                  float* __restrict__ Y) {
  __shared__ unsigned short lX[64 * NF];   // 16 KB
  __shared__ unsigned short lH[64 * NF];   // 16 KB

  const int tid  = threadIdx.x;
  const int lane = tid & 31;
  const int wav  = tid >> 5;
  const int rb   = wav & 3;      // row block 0..3 (16 rows each)
  const int cg   = wav >> 2;     // col group 0..1 (64 cols each)
  const long long rowBaseG = (long long)blockIdx.x * 64;

  __builtin_prefetch(sWa, 0, 0);

  for (int i = tid; i < 64 * NF; i += 256)
    lX[i] = f2bf(X[rowBaseG * NF + i]);
  __syncthreads();

  v8f acc[4];
#pragma unroll
  for (int c = 0; c < 4; ++c) acc[c] = vzero8();

#pragma unroll
  for (int kb = 0; kb < 4; ++kb) {                    // K = 128
    v16bf a = load_a_frag(lX, rb * 16, NF, kb, lane);
#pragma unroll
    for (int ct = 0; ct < 4; ++ct) {
      v16bf bf = load_b_frag(sWa, NF / 16, kb, cg * 4 + ct, lane);
      acc[ct] = WMMA_BF16(a, bf, acc[ct]);
    }
  }

  if constexpr (TWO_LAYER) {
    // bias + tanh -> bf16 hidden in LDS
#pragma unroll
    for (int ct = 0; ct < 4; ++ct) {
      int col = cg * 64 + ct * 16 + (lane & 15);
      float bc = ba[col];
      int rh = (lane >> 4) * 8;
#pragma unroll
      for (int v = 0; v < 8; ++v)
        lH[(rb * 16 + rh + v) * NF + col] = f2bf(tanhf(acc[ct][v] + bc));
    }
    __syncthreads();
#pragma unroll
    for (int c = 0; c < 4; ++c) acc[c] = vzero8();
#pragma unroll
    for (int kb = 0; kb < 4; ++kb) {
      v16bf a = load_a_frag(lH, rb * 16, NF, kb, lane);
#pragma unroll
      for (int ct = 0; ct < 4; ++ct) {
        v16bf bf = load_b_frag(sWb, NF / 16, kb, cg * 4 + ct, lane);
        acc[ct] = WMMA_BF16(a, bf, acc[ct]);
      }
    }
#pragma unroll
    for (int ct = 0; ct < 4; ++ct) {
      int col = cg * 64 + ct * 16 + (lane & 15);
      float bc = bb[col];
      int rh = (lane >> 4) * 8;
#pragma unroll
      for (int v = 0; v < 8; ++v) {
        long long row = rowBaseG + rb * 16 + rh + v;
        Y[row * NF + col] = acc[ct][v] + bc;
      }
    }
  } else {
#pragma unroll
    for (int ct = 0; ct < 4; ++ct) {
      int col = cg * 64 + ct * 16 + (lane & 15);
      float bc = ba ? ba[col] : 0.f;
      int rh = (lane >> 4) * 8;
#pragma unroll
      for (int v = 0; v < 8; ++v) {
        long long row = rowBaseG + rb * 16 + rh + v;
        Y[row * NF + col] = acc[ct][v] + bc;
      }
    }
  }
}

// ---------- fused CFConv + attention: one workgroup per atom ----------
__global__ void cfconv_kernel(const float* __restrict__ initF,        // [8000,128] f32 (ws)
                              const float* __restrict__ rbf,          // [8,1000,64,50]
                              const long long* __restrict__ nl,       // [8,1000,64] int64
                              const unsigned short* __restrict__ sW1, // swz [64,128]
                              const float* __restrict__ b1,
                              const unsigned short* __restrict__ sW2, // swz [128,128]
                              const float* __restrict__ b2,
                              const float* __restrict__ nbrw,         // [128]
                              float* __restrict__ aggOut,             // [8000,128] f32 (ws)
                              float* __restrict__ attnOut) {          // [8000,64] f32 (d_out tail)
  __shared__ unsigned short lA[NM * 64];   // rbf tile bf16, G padded 50->64 (8 KB)
  // lsBuf aliases: first raw f32 rbf staging (12,800 B), later hidden bf16 (16,384 B).
  // Lifetimes are disjoint (separated by a barrier after the bf16 conversion).
  __shared__ char lsBuf[NM * NF * 2];      // 16 KB
  __shared__ float lInit[NM * NF];         // gathered neighbor feats (32 KB)
  __shared__ float sScore[NM];
  __shared__ float sAttn[NM];
  __shared__ float sAgg[NF];
  __shared__ float sNbrw[NF];

  float*          lRaw = (float*)lsBuf;           // [64*50] raw rbf
  unsigned short* lH   = (unsigned short*)lsBuf;  // [64*128] hidden bf16

  const int tid  = threadIdx.x;
  const int lane = tid & 31;
  const int wav  = tid >> 5;
  const int rb   = wav & 3;
  const int cg   = wav >> 2;
  const int atom = blockIdx.x;                 // b*1000 + n
  const int bIdx = atom / NATOM;

  __builtin_prefetch(sW1, 0, 0);
  __builtin_prefetch(sW2, 0, 0);

  // --- async-stage the HBM stream: raw rbf tile (12,800 contiguous bytes) ---
  const float* rbfBase = rbf + (long long)atom * NM * NG;
  for (int i = tid; i < (NM * NG * 4) / 16; i += 256)        // 800 b128 chunks
    async_g2l_b128((char*)lRaw + i * 16, (const char*)rbfBase + i * 16);

  // --- async gather of neighbor init rows (per-lane addressed, L2 hits) ---
  {
    int m = tid >> 2, q = tid & 3;
    int nrow = (int)nl[(long long)atom * NM + m];
    const float* src = initF + ((long long)bIdx * NATOM + nrow) * NF;
#pragma unroll
    for (int i = 0; i < 8; ++i) {
      int c = q * 32 + i * 4;
      async_g2l_b128(&lInit[m * NF + c], src + c);
    }
  }

  if (tid < NM) sScore[tid] = 0.f;
  if (tid < NF) { sAgg[tid] = 0.f; sNbrw[tid] = nbrw[tid]; }

  wait_asynccnt0();
  __syncthreads();

  // convert staged rbf f32 -> bf16, zero-pad G 50->64
  for (int i = tid; i < NM * 64; i += 256) {
    int g = i & 63;
    float v = (g < NG) ? lRaw[(i >> 6) * NG + g] : 0.f;
    lA[i] = f2bf(v);
  }
  __syncthreads();          // lRaw dead from here on; lsBuf reused as lH

  // GEMM1: hidden = tanh(rbf @ W1 + b1), K=64
  v8f acc[4];
#pragma unroll
  for (int c = 0; c < 4; ++c) acc[c] = vzero8();
#pragma unroll
  for (int kb = 0; kb < 2; ++kb) {
    v16bf a = load_a_frag(lA, rb * 16, 64, kb, lane);
#pragma unroll
    for (int ct = 0; ct < 4; ++ct) {
      v16bf bf = load_b_frag(sW1, NF / 16, kb, cg * 4 + ct, lane);
      acc[ct] = WMMA_BF16(a, bf, acc[ct]);
    }
  }
#pragma unroll
  for (int ct = 0; ct < 4; ++ct) {
    int col = cg * 64 + ct * 16 + (lane & 15);
    float bc = b1[col];
    int rh = (lane >> 4) * 8;
#pragma unroll
    for (int v = 0; v < 8; ++v)
      lH[(rb * 16 + rh + v) * NF + col] = f2bf(tanhf(acc[ct][v] + bc));
  }
  __syncthreads();

  // GEMM2: filt = hidden @ W2 + b2, K=128; then conv = filt * gathered init
#pragma unroll
  for (int c = 0; c < 4; ++c) acc[c] = vzero8();
#pragma unroll
  for (int kb = 0; kb < 4; ++kb) {
    v16bf a = load_a_frag(lH, rb * 16, NF, kb, lane);
#pragma unroll
    for (int ct = 0; ct < 4; ++ct) {
      v16bf bf = load_b_frag(sW2, NF / 16, kb, cg * 4 + ct, lane);
      acc[ct] = WMMA_BF16(a, bf, acc[ct]);
    }
  }
#pragma unroll
  for (int ct = 0; ct < 4; ++ct) {
    int col = cg * 64 + ct * 16 + (lane & 15);
    float bc = b2[col];
    float wv = sNbrw[col];
    int rh = (lane >> 4) * 8;
#pragma unroll
    for (int v = 0; v < 8; ++v) {
      int row = rb * 16 + rh + v;
      float conv = (acc[ct][v] + bc) * lInit[row * NF + col];
      acc[ct][v] = conv;                         // keep conv in registers
      atomicAdd(&sScore[row], conv * wv);        // ds_add_f32
    }
  }
  __syncthreads();

  if (tid == 0) {                                // softmax over 64 neighbors
    float mx = -1e30f;
    for (int i = 0; i < NM; ++i) mx = fmaxf(mx, sScore[i]);
    float s = 0.f;
    for (int i = 0; i < NM; ++i) { float e = __expf(sScore[i] - mx); sAttn[i] = e; s += e; }
    float inv = 1.f / s;
    for (int i = 0; i < NM; ++i) sAttn[i] *= inv;
  }
  __syncthreads();

  // agg[f] = sum_m attn[m]*conv[m][f]  (conv still in accumulators)
#pragma unroll
  for (int ct = 0; ct < 4; ++ct) {
    int col = cg * 64 + ct * 16 + (lane & 15);
    int rh = (lane >> 4) * 8;
#pragma unroll
    for (int v = 0; v < 8; ++v) {
      int row = rb * 16 + rh + v;
      atomicAdd(&sAgg[col], sAttn[row] * acc[ct][v]);
    }
  }
  __syncthreads();

  if (tid < NF) aggOut[(long long)atom * NF + tid] = sAgg[tid];
  if (tid < NM) attnOut[(long long)atom * NM + tid] = sAttn[tid];
}

// ---------- launch ----------
extern "C" void kernel_launch(void* const* d_in, const int* in_sizes, int n_in,
                              void* d_out, int out_size, void* d_ws, size_t ws_size,
                              hipStream_t stream) {
  const float*     features = (const float*)d_in[0];
  const float*     rbf      = (const float*)d_in[1];
  const long long* nl       = (const long long*)d_in[2];   // jnp.int64
  const float*     W_init   = (const float*)d_in[3];
  const float*     W1       = (const float*)d_in[4];
  const float*     b1       = (const float*)d_in[5];
  const float*     W2       = (const float*)d_in[6];
  const float*     b2       = (const float*)d_in[7];
  const float*     nbrw     = (const float*)d_in[8];
  const float*     Wo1      = (const float*)d_in[9];
  const float*     bo1      = (const float*)d_in[10];
  const float*     Wo2      = (const float*)d_in[11];
  const float*     bo2      = (const float*)d_in[12];

  // workspace layout (bytes): init f32 | agg f32 | swizzled bf16 weights
  char* ws = (char*)d_ws;
  float*          initBuf = (float*)(ws + 0);                       // 4,096,000 B
  float*          aggBuf  = (float*)(ws + 4096000);                 // 4,096,000 B
  unsigned short* sWinit  = (unsigned short*)(ws + 8192000);        // 32,768 B
  unsigned short* sW1     = (unsigned short*)(ws + 8192000 + 32768);  // 16,384 B
  unsigned short* sW2     = (unsigned short*)(ws + 8192000 + 49152);  // 32,768 B
  unsigned short* sWo1    = (unsigned short*)(ws + 8192000 + 81920);  // 32,768 B
  unsigned short* sWo2    = (unsigned short*)(ws + 8192000 + 114688); // 32,768 B
  // total ws needed: 8,339,456 bytes

  float* outBuf  = (float*)d_out;            // [8000,128]
  float* attnBuf = outBuf + OUT_ELEMS;       // [8000,64]

  // 1) pre-swizzle weights to bf16 fragment order
  swizzle_w_kernel<<<(128 * 128 + 255) / 256, 256, 0, stream>>>(W_init, sWinit, 128, 128, 128);
  swizzle_w_kernel<<<(64 * 128 + 255) / 256, 256, 0, stream>>>(W1, sW1, 64, NG, 128);
  swizzle_w_kernel<<<(128 * 128 + 255) / 256, 256, 0, stream>>>(W2, sW2, 128, 128, 128);
  swizzle_w_kernel<<<(128 * 128 + 255) / 256, 256, 0, stream>>>(Wo1, sWo1, 128, 128, 128);
  swizzle_w_kernel<<<(128 * 128 + 255) / 256, 256, 0, stream>>>(Wo2, sWo2, 128, 128, 128);

  // 2) init = features @ W_init   (125 tiles of 64 rows)
  dense_tile_kernel<false><<<ROWS_TOT / 64, 256, 0, stream>>>(
      features, sWinit, nullptr, nullptr, nullptr, initBuf);

  // 3) fused CFConv + attention, one WG per atom
  cfconv_kernel<<<ROWS_TOT, 256, 0, stream>>>(
      initBuf, rbf, nl, sW1, b1, sW2, b2, nbrw, aggBuf, attnBuf);

  // 4) out = tanh(agg @ Wo1 + bo1) @ Wo2 + bo2
  dense_tile_kernel<true><<<ROWS_TOT / 64, 256, 0, stream>>>(
      aggBuf, sWo1, bo1, sWo2, bo2, outBuf);
}
